// CausalSelfAttention_60713657696818
// MI455X (gfx1250) — matrix operand order repaired and checked by reading, not compiled, not run on hardware
//
#include <hip/hip_runtime.h>
#include <hip/hip_bf16.h>

// ---------------------------------------------------------------------------
// CDNA5 (gfx1250) multi-head attention, bf16 WMMA (16x16x32, f32 accumulate)
// Double-buffered LDS tiles, branchless templated epilogues, 64-key flash
// blocks to amortize softmax/barrier overhead over 16 WMMAs per iteration.
// ---------------------------------------------------------------------------

typedef __attribute__((ext_vector_type(16))) __bf16 v16bf;
typedef __attribute__((ext_vector_type(8)))  float  v8f;

__device__ __forceinline__ v8f wmma_bf16(v16bf a, v16bf b, v8f c) {
  // D = A(16x32) * B(32x16) + C(16x16 f32)
  return __builtin_amdgcn_wmma_f32_16x16x32_bf16(
      /*neg_a=*/false, a, /*neg_b=*/false, b,
      /*c_mod=*/(short)0, c, /*reuse_a=*/false, /*reuse_b=*/false);
}

#define D_MODEL 768
#define SEQ     2048
#define NHEAD   12
#define HDIM    64

// ---------------------------------------------------------------------------
// f32 -> bf16 conversion (vectorized: float4 load, 4x bf16 (8B) store)
// ---------------------------------------------------------------------------
struct bf16x4 { __bf16 a, b, c, d; };

__global__ void cvt_f32_bf16(const float* __restrict__ s,
                             __bf16* __restrict__ d, int n4) {
  int i = blockIdx.x * blockDim.x + threadIdx.x;
  int st = gridDim.x * blockDim.x;
  for (; i < n4; i += st) {
    float4 f = ((const float4*)s)[i];
    bf16x4 o{(__bf16)f.x, (__bf16)f.y, (__bf16)f.z, (__bf16)f.w};
    ((bf16x4*)d)[i] = o;
  }
}

// ---------------------------------------------------------------------------
// Tiled bf16 GEMM + bias:  C[M,N] = A[M,K] * B[K,N] + bias[N]
// Block tile 128x128, 8 waves (4Mx2N), wave tile 32x64, K-step 32.
// Double-buffered LDS; global prefetch overlaps WMMA compute.
// ---------------------------------------------------------------------------
template <bool WRITE_F32>
__global__ __launch_bounds__(256) void gemm_bias_bf16(
    const __bf16* __restrict__ A, const __bf16* __restrict__ Bm,
    const float* __restrict__ bias,
    __bf16* __restrict__ Cb, float* __restrict__ Cf,
    int M, int N, int K) {
  __shared__ __bf16 sA[2][128][32];   // [m][k]             2 x 8 KB
  __shared__ __bf16 sBt[2][128][32];  // transposed: [n][k] 2 x 8 KB

  const int tid  = threadIdx.x;
  const int lane = tid & 31;
  const int wid  = tid >> 5;
  const int ln   = lane & 15;
  const int hi   = lane >> 4;
  const int wm   = wid & 3;   // 4 waves along M (32 rows each)
  const int wn   = wid >> 2;  // 2 waves along N (64 cols each)

  const int n0 = blockIdx.x * 128;
  const int m0 = blockIdx.y * 128;

  v8f acc[2][4] = {};

  // per-thread slices of the cooperative tile loads
  const int ra = tid >> 1, ca = (tid & 1) * 16;   // A: 128 rows x 32 k
  const int rb = tid & 31, cb = (tid >> 5) * 16;  // B: 32 k x 128 n
  const __bf16* gA = A + (size_t)(m0 + ra) * K + ca;
  const __bf16* gB = Bm + (size_t)rb * N + n0 + cb;

  const int nk = K / 32;

  // prologue: stage k-slab 0 in registers
  int4 a0 = *(const int4*)gA;
  int4 a1 = *(const int4*)(gA + 8);
  int4 b0 = *(const int4*)gB;
  int4 b1 = *(const int4*)(gB + 8);

  for (int ks = 0; ks < nk; ++ks) {
    const int cur = ks & 1;
    // commit staged slab to LDS buffer `cur`
    *(int4*)&sA[cur][ra][ca]     = a0;
    *(int4*)&sA[cur][ra][ca + 8] = a1;
    {
      const __bf16* e0 = (const __bf16*)&b0;
      const __bf16* e1 = (const __bf16*)&b1;
#pragma unroll
      for (int j = 0; j < 8; ++j) {
        sBt[cur][cb + j][rb]     = e0[j];
        sBt[cur][cb + 8 + j][rb] = e1[j];
      }
    }
    // prefetch next slab; loads stay in flight across the compute below
    if (ks + 1 < nk) {
      const __bf16* nA = gA + (ks + 1) * 32;
      const __bf16* nB = gB + (size_t)(ks + 1) * 32 * N;
      a0 = *(const int4*)nA;  a1 = *(const int4*)(nA + 8);
      b0 = *(const int4*)nB;  b1 = *(const int4*)(nB + 8);
    }
    __syncthreads();

    // --- fragments + 8 WMMA -----------------------------------------------
    v16bf afrag[2], bfrag[4];
#pragma unroll
    for (int im = 0; im < 2; ++im) {
      const __bf16* p = &sA[cur][wm * 32 + im * 16 + ln][0];
      ((int4*)&afrag[im])[0] = *(const int4*)(p + hi * 8);
      ((int4*)&afrag[im])[1] = *(const int4*)(p + 16 + hi * 8);
    }
#pragma unroll
    for (int jn = 0; jn < 4; ++jn) {
      const __bf16* p = &sBt[cur][wn * 64 + jn * 16 + ln][0];
      ((int4*)&bfrag[jn])[0] = *(const int4*)(p + hi * 8);
      ((int4*)&bfrag[jn])[1] = *(const int4*)(p + 16 + hi * 8);
    }
#pragma unroll
    for (int im = 0; im < 2; ++im)
#pragma unroll
      for (int jn = 0; jn < 4; ++jn)
        acc[im][jn] = wmma_bf16(afrag[im], bfrag[jn], acc[im][jn]);

    __syncthreads();
  }

  // --- epilogue: bias + store (branchless, template-selected path) --------
#pragma unroll
  for (int im = 0; im < 2; ++im)
#pragma unroll
    for (int jn = 0; jn < 4; ++jn) {
      const int col = n0 + wn * 64 + jn * 16 + ln;
      const float bv = bias[col];
#pragma unroll
      for (int v = 0; v < 8; ++v) {
        const int row = m0 + wm * 32 + im * 16 + v + 8 * hi;
        const float val = acc[im][jn][v] + bv;
        if constexpr (WRITE_F32) Cf[(size_t)row * N + col] = val;
        else                     Cb[(size_t)row * N + col] = (__bf16)val;
      }
    }
}

// ---------------------------------------------------------------------------
// Flash attention (no mask). One block = 128 query rows of one (b,h);
// 8 waves x 16 rows. K/V streamed in 64-key blocks, double-buffered LDS.
// Per iteration: 8 WMMA QK^T + 8 WMMA PV; softmax reductions amortized.
// Q,K,V stored as bf16 [B*S, 768] with head h in cols h*64..h*64+63.
// ---------------------------------------------------------------------------
__global__ __launch_bounds__(256) void attn_flash(
    const __bf16* __restrict__ Qb, const __bf16* __restrict__ Kb,
    const __bf16* __restrict__ Vb, __bf16* __restrict__ Ob) {
  __shared__ __bf16 sK[2][64][64];   // [key][d]          2 x 8 KB
  __shared__ __bf16 sVt[2][64][64];  // [d][key]          2 x 8 KB
  __shared__ __bf16 sP[8][16][64];   // per-wave P stage  16 KB

  const int tid  = threadIdx.x;
  const int lane = tid & 31;
  const int w    = tid >> 5;
  const int ln   = lane & 15;
  const int hi   = lane >> 4;
  const int h    = blockIdx.y;
  const int b    = blockIdx.z;
  const int colH = h * HDIM;
  const int qrow = b * SEQ + blockIdx.x * 128 + w * 16 + ln;  // A-frag row

  // Q fragments: 2 chunks of 32 over Dh=64
  v16bf qa[2];
#pragma unroll
  for (int c = 0; c < 2; ++c) {
    const __bf16* p = Qb + (size_t)qrow * D_MODEL + colH + c * 32;
    ((int4*)&qa[c])[0] = *(const int4*)(p + hi * 8);
    ((int4*)&qa[c])[1] = *(const int4*)(p + 16 + hi * 8);
  }

  v8f o[4] = {};
  float m_[8], l_[8];
#pragma unroll
  for (int v = 0; v < 8; ++v) { m_[v] = -3.0e30f; l_[v] = 0.0f; }

  const float scale = 0.125f;  // 1/sqrt(64)

  // per-thread slice of the cooperative K/V tile load (64 keys x 64 d)
  const int key = tid & 63, dgrp = (tid >> 6) * 16;
  const size_t gstep = (size_t)64 * D_MODEL;
  const __bf16* gK = Kb + (size_t)(b * SEQ + key) * D_MODEL + colH + dgrp;
  const __bf16* gV = Vb + (size_t)(b * SEQ + key) * D_MODEL + colH + dgrp;

  const int nit = SEQ / 64;

  // prologue: stage key-block 0
  int4 k0v = *(const int4*)gK;
  int4 k1v = *(const int4*)(gK + 8);
  int4 v0v = *(const int4*)gV;
  int4 v1v = *(const int4*)(gV + 8);

  for (int it = 0; it < nit; ++it) {
    const int cur = it & 1;
    // commit staged K/V block (V transposed) to LDS buffer `cur`
    *(int4*)&sK[cur][key][dgrp]     = k0v;
    *(int4*)&sK[cur][key][dgrp + 8] = k1v;
    {
      const __bf16* e0 = (const __bf16*)&v0v;
      const __bf16* e1 = (const __bf16*)&v1v;
#pragma unroll
      for (int j = 0; j < 8; ++j) {
        sVt[cur][dgrp + j][key]     = e0[j];
        sVt[cur][dgrp + 8 + j][key] = e1[j];
      }
    }
    // prefetch next key-block; overlaps with the compute below
    if (it + 1 < nit) {
      const __bf16* nK = gK + (size_t)(it + 1) * gstep;
      const __bf16* nV = gV + (size_t)(it + 1) * gstep;
      k0v = *(const int4*)nK;  k1v = *(const int4*)(nK + 8);
      v0v = *(const int4*)nV;  v1v = *(const int4*)(nV + 8);
    }
    __syncthreads();

    // --- scores: S(16x64) = Q(16x64) * K^T(64x64), 8 WMMA -----------------
    v8f s[4] = {};
#pragma unroll
    for (int c = 0; c < 2; ++c)
#pragma unroll
      for (int kt = 0; kt < 4; ++kt) {
        const __bf16* p = &sK[cur][kt * 16 + ln][c * 32];
        v16bf kbf;
        ((int4*)&kbf)[0] = *(const int4*)(p + hi * 8);
        ((int4*)&kbf)[1] = *(const int4*)(p + 16 + hi * 8);
        s[kt] = wmma_bf16(qa[c], kbf, s[kt]);
      }

    // --- online softmax across the 64 new columns -------------------------
    float alpha[8];
#pragma unroll
    for (int v = 0; v < 8; ++v) {
      float a0 = s[0][v] * scale, a1 = s[1][v] * scale;
      float a2 = s[2][v] * scale, a3 = s[3][v] * scale;
      float mx = fmaxf(fmaxf(a0, a1), fmaxf(a2, a3));
      mx = fmaxf(mx, __shfl_xor(mx, 1, 32));
      mx = fmaxf(mx, __shfl_xor(mx, 2, 32));
      mx = fmaxf(mx, __shfl_xor(mx, 4, 32));
      mx = fmaxf(mx, __shfl_xor(mx, 8, 32));
      float mnew = fmaxf(m_[v], mx);
      float p0 = __expf(a0 - mnew);
      float p1 = __expf(a1 - mnew);
      float p2 = __expf(a2 - mnew);
      float p3 = __expf(a3 - mnew);
      float r  = (p0 + p1) + (p2 + p3);
      r += __shfl_xor(r, 1, 32);
      r += __shfl_xor(r, 2, 32);
      r += __shfl_xor(r, 4, 32);
      r += __shfl_xor(r, 8, 32);
      alpha[v] = __expf(m_[v] - mnew);
      m_[v] = mnew;
      l_[v] = l_[v] * alpha[v] + r;
      // P staging (row = v+8*hi matches C-layout; col = key within block)
      sP[w][v + 8 * hi][ln]      = (__bf16)p0;
      sP[w][v + 8 * hi][16 + ln] = (__bf16)p1;
      sP[w][v + 8 * hi][32 + ln] = (__bf16)p2;
      sP[w][v + 8 * hi][48 + ln] = (__bf16)p3;
    }
    // rescale running output
#pragma unroll
    for (int jt = 0; jt < 4; ++jt)
#pragma unroll
      for (int v = 0; v < 8; ++v) o[jt][v] *= alpha[v];

    // intra-wave LDS store->load handoff for the P transpose
    asm volatile("s_wait_dscnt 0" ::: "memory");

    // --- reload P as 2 A fragments (16x64 bf16) ---------------------------
    v16bf pa[2];
#pragma unroll
    for (int c = 0; c < 2; ++c) {
      const __bf16* pp = &sP[w][ln][c * 32];
      ((int4*)&pa[c])[0] = *(const int4*)(pp + hi * 8);
      ((int4*)&pa[c])[1] = *(const int4*)(pp + 16 + hi * 8);
    }
    // --- O(16x64) += P(16x64) * V(64x64), 8 WMMA --------------------------
#pragma unroll
    for (int jt = 0; jt < 4; ++jt)
#pragma unroll
      for (int c = 0; c < 2; ++c) {
        const __bf16* pv = &sVt[cur][jt * 16 + ln][c * 32];
        v16bf vbf;
        ((int4*)&vbf)[0] = *(const int4*)(pv + hi * 8);
        ((int4*)&vbf)[1] = *(const int4*)(pv + 16 + hi * 8);
        o[jt] = wmma_bf16(pa[c], vbf, o[jt]);
      }

    __syncthreads();
  }

  // --- epilogue: normalize and store bf16 for the output projection -------
#pragma unroll
  for (int jt = 0; jt < 4; ++jt)
#pragma unroll
    for (int v = 0; v < 8; ++v) {
      int row = b * SEQ + blockIdx.x * 128 + w * 16 + v + 8 * hi;
      int col = colH + jt * 16 + ln;
      Ob[(size_t)row * D_MODEL + col] = (__bf16)(o[jt][v] / l_[v]);
    }
}

// ---------------------------------------------------------------------------
// Launch
// ---------------------------------------------------------------------------
extern "C" void kernel_launch(void* const* d_in, const int* in_sizes, int n_in,
                              void* d_out, int out_size, void* d_ws, size_t ws_size,
                              hipStream_t stream) {
  (void)in_sizes; (void)n_in; (void)out_size; (void)ws_size;
  const float* x  = (const float*)d_in[0];
  const float* Wq = (const float*)d_in[1];
  const float* bq = (const float*)d_in[2];
  const float* Wk = (const float*)d_in[3];
  const float* bk = (const float*)d_in[4];
  const float* Wv = (const float*)d_in[5];
  const float* bv = (const float*)d_in[6];
  const float* Wo = (const float*)d_in[7];
  const float* bo = (const float*)d_in[8];
  float* out = (float*)d_out;

  const size_t SZ_X = (size_t)4096 * D_MODEL;     // B*S x 768
  const size_t SZ_W = (size_t)D_MODEL * D_MODEL;

  __bf16* xb  = (__bf16*)d_ws;
  __bf16* wqb = xb + SZ_X;
  __bf16* wkb = wqb + SZ_W;
  __bf16* wvb = wkb + SZ_W;
  __bf16* wob = wvb + SZ_W;
  __bf16* qb  = wob + SZ_W;
  __bf16* kb  = qb + SZ_X;
  __bf16* vb  = kb + SZ_X;
  __bf16* ab  = vb + SZ_X;

  // 1) f32 -> bf16 conversions
  cvt_f32_bf16<<<512, 256, 0, stream>>>(x,  xb,  (int)(SZ_X / 4));
  cvt_f32_bf16<<<256, 256, 0, stream>>>(Wq, wqb, (int)(SZ_W / 4));
  cvt_f32_bf16<<<256, 256, 0, stream>>>(Wk, wkb, (int)(SZ_W / 4));
  cvt_f32_bf16<<<256, 256, 0, stream>>>(Wv, wvb, (int)(SZ_W / 4));
  cvt_f32_bf16<<<256, 256, 0, stream>>>(Wo, wob, (int)(SZ_W / 4));

  // 2) QKV projections: [4096,768] = xb @ W + b  (bf16 out)
  dim3 ggrid(D_MODEL / 128, 4096 / 128);
  gemm_bias_bf16<false><<<ggrid, 256, 0, stream>>>(xb, wqb, bq, qb, nullptr,
                                                   4096, D_MODEL, D_MODEL);
  gemm_bias_bf16<false><<<ggrid, 256, 0, stream>>>(xb, wkb, bk, kb, nullptr,
                                                   4096, D_MODEL, D_MODEL);
  gemm_bias_bf16<false><<<ggrid, 256, 0, stream>>>(xb, wvb, bv, vb, nullptr,
                                                   4096, D_MODEL, D_MODEL);

  // 3) attention (flash, no mask): grid = (S/128, H, B)
  dim3 agrid(SEQ / 128, NHEAD, 2);
  attn_flash<<<agrid, 256, 0, stream>>>(qb, kb, vb, ab);

  // 4) output projection: f32 out = ab @ Wo + bo
  gemm_bias_bf16<true><<<ggrid, 256, 0, stream>>>(ab, wob, bo, nullptr, out,
                                                  4096, D_MODEL, D_MODEL);
}
